// LatentAlignmentLoss_85057532330126
// MI455X (gfx1250) — matrix operand order repaired and checked by reading, not compiled, not run on hardware
//
#include <hip/hip_runtime.h>
#include <hip/hip_fp16.h>

// ---------------------------------------------------------------------------
// LatentAlignmentLoss for MI455X (gfx1250, wave32, WMMA).
//
// Both loss terms reduce to Gram matrices:
//   S = Zn Zn^T  (8192x8192 from 8192x256)  -> fused exp/log/sum epilogue
//   G = Bs Bs^T  (8192x8192 from 8192x32)   -> fused top-5 NN epilogue
// Neither Gram matrix is materialized. Main GEMM = 34 GFLOP on f16 WMMA
// (v_wmma_f32_16x16x32_f16, K-loop of 8, B tile double-buffered); the 134M
// v_exp_f32 epilogue ops are the throughput limiter, HBM traffic ~16 MB
// (Zh re-reads are served out of the 192 MB L2).
// ---------------------------------------------------------------------------

typedef __attribute__((ext_vector_type(16))) _Float16 v16h;
typedef __attribute__((ext_vector_type(8)))  _Float16 v8h;
typedef __attribute__((ext_vector_type(8)))  float    v8f;

#define NROWS   8192
#define DIM     256
#define BDIM    32
#define TAU_INV 10.0f
#define LOG2E   1.4426950408889634f
#define LN2     0.6931471805599453f

__device__ __forceinline__ unsigned long long u64min(unsigned long long a,
                                                     unsigned long long b) {
  return a < b ? a : b;
}
__device__ __forceinline__ unsigned long long u64max(unsigned long long a,
                                                     unsigned long long b) {
  return a < b ? b : a;
}
__device__ __forceinline__ unsigned long long shfl_xor_u64(unsigned long long x,
                                                           int mask) {
  unsigned lo = (unsigned)x, hi = (unsigned)(x >> 32);
  lo = __shfl_xor(lo, mask, 32);
  hi = __shfl_xor(hi, mask, 32);
  return ((unsigned long long)hi << 32) | lo;
}

// --------------------------- accumulator init ------------------------------
__global__ void init_acc_kernel(float* acc) {
  if (threadIdx.x < 2) acc[threadIdx.x] = 0.0f;
}

// --------------------- row L2-normalize z, cast to f16 ---------------------
__global__ __launch_bounds__(256) void normalize_z_kernel(
    const float* __restrict__ z, _Float16* __restrict__ Zh) {
  const int row  = blockIdx.x * 8 + (threadIdx.x >> 5);
  const int lane = threadIdx.x & 31;
  const float* zr = z + (size_t)row * DIM;
  float vals[8];
  float ss = 0.0f;
#pragma unroll
  for (int k = 0; k < 8; ++k) {
    vals[k] = zr[lane + 32 * k];
    ss += vals[k] * vals[k];
  }
#pragma unroll
  for (int m = 1; m < 32; m <<= 1) ss += __shfl_xor(ss, m, 32);
  const float rn = rsqrtf(ss);
  _Float16* o = Zh + (size_t)row * DIM;
#pragma unroll
  for (int k = 0; k < 8; ++k) o[lane + 32 * k] = (_Float16)(vals[k] * rn);
}

// ------------- binding_scores -> f16 + per-row squared norms ---------------
__global__ __launch_bounds__(256) void prep_binding_kernel(
    const float* __restrict__ bs, _Float16* __restrict__ Bh,
    float* __restrict__ sqB) {
  const int row  = blockIdx.x * 8 + (threadIdx.x >> 5);
  const int lane = threadIdx.x & 31;
  const float v = bs[(size_t)row * BDIM + lane];
  Bh[(size_t)row * BDIM + lane] = (_Float16)v;
  float ss = v * v;
#pragma unroll
  for (int m = 1; m < 32; m <<= 1) ss += __shfl_xor(ss, m, 32);
  if (lane == 0) sqB[row] = ss;
}

// -------- top-5 nearest neighbours of binding rows via WMMA Gram -----------
// Block = 16-row strip x all 8192 cols; 8 waves, wave w owns col tiles w+8t.
// Per thread: sorted top-5 u64 keys per owned row (d2 bits << 32 | col).
// Diagonal handled branchlessly (key poisoned to UINT64_MAX).
__global__ __launch_bounds__(256) void topk_pos_kernel(
    const _Float16* __restrict__ Bh, const float* __restrict__ sqB,
    int* __restrict__ pos) {
  __shared__ unsigned long long cand[8 * 16 * 5];  // [wave][row][5]

  const int strip = blockIdx.x;
  const int tid   = threadIdx.x;
  const int wave  = tid >> 5;
  const int lane  = tid & 31;
  const int half  = lane >> 4;
  const int l16   = lane & 15;
  const unsigned long long KMAX = ~0ull;

  // A tile (16x32 f16), contiguous layout: lane<16 -> K 0-15; lane>=16 -> K 16-31
  v16h a;
  {
    const _Float16* arow = Bh + (size_t)(strip * 16 + l16) * BDIM + half * 16;
    v8h lo = *(const v8h*)(arow);
    v8h hi = *(const v8h*)(arow + 8);
#pragma unroll
    for (int e = 0; e < 8; ++e) { a[e] = lo[e]; a[e + 8] = hi[e]; }
  }
  float sqLoc[8];
#pragma unroll
  for (int v = 0; v < 8; ++v) sqLoc[v] = sqB[strip * 16 + v + 8 * half];

  unsigned long long top[8][5];
#pragma unroll
  for (int v = 0; v < 8; ++v)
#pragma unroll
    for (int k = 0; k < 5; ++k) top[v][k] = KMAX;

  for (int t = 0; t < 64; ++t) {
    const int jt  = wave + 8 * t;
    const int col = jt * 16 + l16;
    const _Float16* bcol = Bh + (size_t)col * BDIM + half * 16;
    v16h b = *(const v16h*)bcol;
    v8f c = {};
    c = __builtin_amdgcn_wmma_f32_16x16x32_f16(false, a, false, b,
                                               (short)0, c, false, false);
    const float sqc = sqB[col];
#pragma unroll
    for (int v = 0; v < 8; ++v) {
      const int row = strip * 16 + v + 8 * half;
      const float d2 = fmaxf(sqLoc[v] + sqc - 2.0f * c[v], 0.0f);
      unsigned long long key =
          ((unsigned long long)__float_as_uint(d2) << 32) | (unsigned)col;
      key = (col == row) ? KMAX : key;  // self excluded, branchless
      if (key < top[v][4]) {            // usually false -> cheap
        top[v][4] = key;
#pragma unroll
        for (int i = 4; i > 0; --i) {
          unsigned long long lo2 = u64min(top[v][i - 1], top[v][i]);
          unsigned long long hi2 = u64max(top[v][i - 1], top[v][i]);
          top[v][i - 1] = lo2;
          top[v][i]     = hi2;
        }
      }
    }
  }

  // 16-lane min-extract merge (per half => per row), 5 rounds each row
#pragma unroll
  for (int v = 0; v < 8; ++v) {
    unsigned long long res = KMAX;
#pragma unroll
    for (int r = 0; r < 5; ++r) {
      unsigned long long c0 = top[v][0];
      unsigned long long m  = c0;
      m = u64min(m, shfl_xor_u64(m, 1));
      m = u64min(m, shfl_xor_u64(m, 2));
      m = u64min(m, shfl_xor_u64(m, 4));
      m = u64min(m, shfl_xor_u64(m, 8));
      if (l16 == r) res = m;
      const bool owner = (c0 == m);  // unique: col encodes lane identity
#pragma unroll
      for (int i = 0; i < 4; ++i) top[v][i] = owner ? top[v][i + 1] : top[v][i];
      top[v][4] = owner ? KMAX : top[v][4];
    }
    if (l16 < 5) cand[((wave * 16) + v + 8 * half) * 5 + l16] = res;
  }
  __syncthreads();

  // merge 8 wave-lists per row, pick one of top-5 (deterministic hash as
  // stand-in for jax.random.randint(key 42) -- indices are grad-constant)
  if (tid < 16) {
    unsigned long long best[5] = {KMAX, KMAX, KMAX, KMAX, KMAX};
    for (int w = 0; w < 8; ++w)
#pragma unroll
      for (int k = 0; k < 5; ++k) {
        unsigned long long key = cand[(w * 16 + tid) * 5 + k];
        if (key < best[4]) {
          best[4] = key;
#pragma unroll
          for (int i = 4; i > 0; --i) {
            unsigned long long lo2 = u64min(best[i - 1], best[i]);
            unsigned long long hi2 = u64max(best[i - 1], best[i]);
            best[i - 1] = lo2;
            best[i]     = hi2;
          }
        }
      }
    const int r = strip * 16 + tid;
    const unsigned choice = (((unsigned)r * 2654435761u) >> 7) % 5u;
    unsigned long long sel = best[0];
    if (choice == 1) sel = best[1];
    if (choice == 2) sel = best[2];
    if (choice == 3) sel = best[3];
    if (choice == 4) sel = best[4];
    pos[r] = (int)(unsigned)(sel & 0xffffffffu);
  }
}

// ---------------- fused similarity GEMM + InfoNCE/uniformity ---------------
// Block = 16-row strip x all 8192 cols. A (16x256) resident in 64 VGPRs.
// B tiles double-buffered: loads for tile t+1 issue before the 8 WMMAs and
// exp epilogue of tile t, so L2 latency hides under matrix + trans work.
// Epilogue is branchless: pos_sim accumulated via cndmask-select.
__global__ __launch_bounds__(256) void sim_fused_kernel(
    const _Float16* __restrict__ Zh, const int* __restrict__ pos,
    float* __restrict__ acc /* [0]=info sum, [1]=unif sum */) {
  __shared__ float denomLDS[16];
  __shared__ float posSimLDS[16];
  __shared__ float unifLDS;
  __shared__ int   posLDS[16];

  const int strip = blockIdx.x;
  const int tid   = threadIdx.x;
  const int wave  = tid >> 5;
  const int lane  = tid & 31;
  const int half  = lane >> 4;
  const int l16   = lane & 15;

  if (tid < 16) {
    denomLDS[tid]  = 0.0f;
    posSimLDS[tid] = 0.0f;
    posLDS[tid]    = pos[strip * 16 + tid];
    if (tid == 0) unifLDS = 0.0f;
  }
  __syncthreads();

  int posv[8];
#pragma unroll
  for (int v = 0; v < 8; ++v) posv[v] = posLDS[v + 8 * half];

  // A tile: rows strip*16..+15, K=0..255, contiguous K (half h: K 16h..16h+15 of each 32-block), register-resident.
  v16h A8[8];
  {
    const _Float16* arow = Zh + (size_t)(strip * 16 + l16) * DIM + half * 16;
#pragma unroll
    for (int kb = 0; kb < 8; ++kb) {
      v8h lo = *(const v8h*)(arow + kb * 32);
      v8h hi = *(const v8h*)(arow + kb * 32 + 8);
#pragma unroll
      for (int e = 0; e < 8; ++e) { A8[kb][e] = lo[e]; A8[kb][e + 8] = hi[e]; }
    }
  }

  float d[8], ps[8];
#pragma unroll
  for (int v = 0; v < 8; ++v) { d[v] = 0.0f; ps[v] = 0.0f; }
  float u = 0.0f;

  // ---- software-pipelined column-tile sweep (double-buffered B) ----
  const size_t bstride = (size_t)DIM;  // halves per row
  v16h Bcur[8], Bnxt[8];
  {
    const int col0 = wave * 16 + l16;
    const _Float16* bcol = Zh + (size_t)col0 * bstride + half * 16;
#pragma unroll
    for (int kb = 0; kb < 8; ++kb) Bcur[kb] = *(const v16h*)(bcol + kb * 32);
  }

  for (int t = 0; t < 64; ++t) {
    const int jt  = wave + 8 * t;  // current column tile
    const int col = jt * 16 + l16;
    if (t < 63) {                  // prefetch next tile's B into 2nd buffer
      const int coln = (jt + 8) * 16 + l16;
      const _Float16* bcol = Zh + (size_t)coln * bstride + half * 16;
#pragma unroll
      for (int kb = 0; kb < 8; ++kb) Bnxt[kb] = *(const v16h*)(bcol + kb * 32);
    }

    v8f c = {};
#pragma unroll
    for (int kb = 0; kb < 8; ++kb)
      c = __builtin_amdgcn_wmma_f32_16x16x32_f16(false, A8[kb], false, Bcur[kb],
                                                 (short)0, c, false, false);

    // epilogue: sim s in C layout (VGPR v -> row v+8*half, lane&15 -> col)
#pragma unroll
    for (int v = 0; v < 8; ++v) {
      const float s = c[v];
      d[v] += __builtin_amdgcn_exp2f(s * (TAU_INV * LOG2E));     // exp(s/tau)
      // exp(-2*max(0, 2-2s)) = exp2(min(0, 4(s-1)*log2e))
      u += __builtin_amdgcn_exp2f(fminf(0.0f, (s - 1.0f) * (4.0f * LOG2E)));
      ps[v] += (col == posv[v]) ? s : 0.0f;  // branchless capture
    }

#pragma unroll
    for (int kb = 0; kb < 8; ++kb) Bcur[kb] = Bnxt[kb];
  }

  // per-row reductions across the 16 lanes of each half
#pragma unroll
  for (int v = 0; v < 8; ++v) {
    float x = d[v], p = ps[v];
    x += __shfl_xor(x, 1, 32);
    x += __shfl_xor(x, 2, 32);
    x += __shfl_xor(x, 4, 32);
    x += __shfl_xor(x, 8, 32);
    p += __shfl_xor(p, 1, 32);
    p += __shfl_xor(p, 2, 32);
    p += __shfl_xor(p, 4, 32);
    p += __shfl_xor(p, 8, 32);
    if (l16 == 0) {
      atomicAdd(&denomLDS[v + 8 * half], x);
      atomicAdd(&posSimLDS[v + 8 * half], p);  // only one wave adds nonzero
    }
  }
  float uu = u;
#pragma unroll
  for (int m = 1; m < 32; m <<= 1) uu += __shfl_xor(uu, m, 32);
  if (lane == 0) atomicAdd(&unifLDS, uu);
  __syncthreads();

  if (tid == 0) {
    float infoPart = 0.0f;
#pragma unroll
    for (int r = 0; r < 16; ++r)
      infoPart += __builtin_amdgcn_logf(denomLDS[r] + 1e-8f) * LN2 -
                  posSimLDS[r] * TAU_INV;
    atomicAdd(&acc[0], infoPart);
    atomicAdd(&acc[1], unifLDS);
  }
}

// ------------------------------- finalize ----------------------------------
__global__ void finalize_kernel(const float* __restrict__ acc,
                                float* __restrict__ out) {
  if (threadIdx.x == 0) {
    const float info = acc[0] / (float)NROWS;
    const float unif =
        __builtin_amdgcn_logf(acc[1] / ((float)NROWS * (float)NROWS) + 1e-8f) *
        LN2;
    out[0] = info + 0.1f * unif;
  }
}

// ------------------------------- launcher ----------------------------------
extern "C" void kernel_launch(void* const* d_in, const int* in_sizes, int n_in,
                              void* d_out, int out_size, void* d_ws,
                              size_t ws_size, hipStream_t stream) {
  const float* z  = (const float*)d_in[0];  // [8192,256] f32
  const float* bs = (const float*)d_in[1];  // [8192,32]  f32
  float* out = (float*)d_out;
  (void)in_sizes; (void)n_in; (void)out_size; (void)ws_size;

  char* ws = (char*)d_ws;
  _Float16* Zh  = (_Float16*)(ws);                              // 4 MiB
  _Float16* Bh  = (_Float16*)(ws + (size_t)NROWS * DIM * 2);    // 512 KiB
  float*    sqB = (float*)(ws + (size_t)NROWS * DIM * 2 +
                           (size_t)NROWS * BDIM * 2);           // 32 KiB
  int*      pos = (int*)((char*)sqB + (size_t)NROWS * 4);       // 32 KiB
  float*    acc = (float*)((char*)pos + (size_t)NROWS * 4);     // 8 B

  init_acc_kernel<<<1, 64, 0, stream>>>(acc);
  normalize_z_kernel<<<NROWS / 8, 256, 0, stream>>>(z, Zh);
  prep_binding_kernel<<<NROWS / 8, 256, 0, stream>>>(bs, Bh, sqB);
  topk_pos_kernel<<<NROWS / 16, 256, 0, stream>>>(Bh, sqB, pos);
  sim_fused_kernel<<<NROWS / 16, 256, 0, stream>>>(Zh, pos, acc);
  finalize_kernel<<<1, 32, 0, stream>>>(acc, out);
}
